// Encoder1DBlock_80092550136038
// MI455X (gfx1250) — matrix operand order repaired and checked by reading, not compiled, not run on hardware
//
#include <hip/hip_runtime.h>
#include <hip/hip_bf16.h>

// ---------------------------------------------------------------------------
// Types / helpers
// ---------------------------------------------------------------------------
typedef __attribute__((ext_vector_type(16))) __bf16 bf16x16;
typedef __attribute__((ext_vector_type(8)))  float  floatx8;
typedef __attribute__((ext_vector_type(4)))  unsigned int uint32x4;
typedef __attribute__((ext_vector_type(8)))  int    int32x8;
typedef __attribute__((ext_vector_type(4)))  int    int32x4;
typedef unsigned short u16;
typedef unsigned int   u32;
typedef unsigned long long u64;

union FragBF { bf16x16 v; u32 u[8]; };

__device__ __forceinline__ u16 f2bf(float f) {
    u32 u = __builtin_bit_cast(u32, f);
    u += 0x7FFFu + ((u >> 16) & 1u);          // round-to-nearest-even
    return (u16)(u >> 16);
}

#if __has_builtin(__builtin_amdgcn_tensor_load_to_lds) && \
    __has_builtin(__builtin_amdgcn_s_wait_tensorcnt)
#define USE_TDM 1
#else
#define USE_TDM 0
#endif

#if USE_TDM
// ---------------------------------------------------------------------------
// Tensor Data Mover: 2-D tile load global->LDS with LDS row padding.
// Descriptor layout per CDNA5 ISA sec 8.3-8.6:
//   group0: count=1 | lds_addr[63:32] | global_addr[120:64] | type=2
//   group1: data_size=1(2B) | pad_enable | pad_interval | pad_amount |
//           tensor_dim0/1 | tile_dim0/1 | tensor_dim0_stride (48b)
// pad_interval code c -> pad after 2^(c+1) DWORDs; pad_amount code a -> a+1 DWORDs.
// ---------------------------------------------------------------------------
__device__ __forceinline__ void tdm_load_2d(u32 lds_off, const void* gptr,
                                            u32 tensor_d0, u32 tensor_d1,
                                            u64 stride0_elems,
                                            u32 tile_d0, u32 tile_d1,
                                            u32 pad_interval_code,
                                            u32 pad_amount_code) {
    const u64 ga = (u64)(size_t)gptr;
    uint32x4 g0;
    g0.x = 1u;                                        // count=1, user mode
    g0.y = lds_off;                                   // LDS byte address
    g0.z = (u32)ga;                                   // global addr [95:64]
    g0.w = (u32)((ga >> 32) & 0x01FFFFFFu) | (2u << 30);  // addr hi | type=2
    int32x8 g1;
    g1[0] = (int)((1u << 16)                          // data_size = 2 bytes
                  | (1u << 20)                        // pad_enable
                  | (pad_interval_code << 22)
                  | (pad_amount_code << 25));
    g1[1] = (int)((tensor_d0 & 0xFFFFu) << 16);       // tensor_dim0 lo
    g1[2] = (int)(((tensor_d0 >> 16) & 0xFFFFu) | ((tensor_d1 & 0xFFFFu) << 16));
    g1[3] = (int)(((tensor_d1 >> 16) & 0xFFFFu) | (tile_d0 << 16));
    g1[4] = (int)(tile_d1 & 0xFFFFu);                 // tile_dim1, tile_dim2=0
    g1[5] = (int)(u32)(stride0_elems & 0xFFFFFFFFull);
    g1[6] = (int)(u32)((stride0_elems >> 32) & 0xFFFFull);
    g1[7] = 0;
    const int32x4 gz = {0, 0, 0, 0};
#if defined(__clang_major__) && __clang_major__ >= 23
    const int32x8 gz8 = {0, 0, 0, 0, 0, 0, 0, 0};
    __builtin_amdgcn_tensor_load_to_lds(g0, g1, gz, gz, gz8, 0);
#else
    __builtin_amdgcn_tensor_load_to_lds(g0, g1, gz, gz, 0);
#endif
}
#endif  // USE_TDM

// ---------------------------------------------------------------------------
// fp32 -> bf16 conversion (weights)
// ---------------------------------------------------------------------------
__global__ __launch_bounds__(256) void cvt_f32_bf16(const float* __restrict__ in,
                                                    u16* __restrict__ out, int n) {
    int i = blockIdx.x * 256 + threadIdx.x;
    if (i < n) out[i] = f2bf(in[i]);
}

// ---------------------------------------------------------------------------
// LayerNorm: fp32 [rows, D] -> bf16 [rows, D].  One block (8 waves) per row.
// ---------------------------------------------------------------------------
__global__ __launch_bounds__(256) void ln_kernel(const float* __restrict__ x,
                                                 const float* __restrict__ g,
                                                 const float* __restrict__ bta,
                                                 u16* __restrict__ out, int D) {
    const int row  = blockIdx.x;
    const int tid  = threadIdx.x;
    const int lane = tid & 31, wave = tid >> 5;

    const float4 v = ((const float4*)(x + (size_t)row * D))[tid];
    float s  = v.x + v.y + v.z + v.w;
    float ss = v.x * v.x + v.y * v.y + v.z * v.z + v.w * v.w;
    #pragma unroll
    for (int m = 16; m >= 1; m >>= 1) {
        s  += __shfl_xor(s,  m, 32);
        ss += __shfl_xor(ss, m, 32);
    }
    __shared__ float rs[8], rss[8];
    if (lane == 0) { rs[wave] = s; rss[wave] = ss; }
    __syncthreads();
    float ts = 0.f, tss = 0.f;
    #pragma unroll
    for (int w = 0; w < 8; ++w) { ts += rs[w]; tss += rss[w]; }
    const float mu   = ts / (float)D;
    const float var  = tss / (float)D - mu * mu;
    const float rstd = rsqrtf(var + 1e-5f);

    const int c0 = tid * 4;
    u16* o = out + (size_t)row * D + c0;
    const float e[4] = {v.x, v.y, v.z, v.w};
    #pragma unroll
    for (int j = 0; j < 4; ++j)
        o[j] = f2bf((e[j] - mu) * rstd * g[c0 + j] + bta[c0 + j]);
}

// ---------------------------------------------------------------------------
// WMMA bf16 GEMM:  C[M,N] = A[M,K] * B[K,N]  (+ epilogue)
//   EPI 0: store bf16 | EPI 1: +bias, exact GELU, bf16 | EPI 2: +bias,+resid, f32
// Block tile 128x128x32; 8 waves (4M x 2N); 8x v_wmma_f32_16x16x32_bf16 per
// wave per K-step.  Operand staging via TDM (double buffered, wave-0 issued,
// s_wait_tensorcnt) with manual global->LDS fallback.
// ---------------------------------------------------------------------------
template <int EPI>
__global__ __launch_bounds__(256) void gemm_bf16_kernel(
    const u16* __restrict__ A, const u16* __restrict__ Bm,
    const float* __restrict__ bias, const float* __restrict__ resid,
    u16* __restrict__ outH, float* __restrict__ outF,
    int M, int N, int K) {
    constexpr int BM = 128, BN = 128, BK = 32;
    constexpr int ASTR = BK + 8;   // 40  (64B rows + 16B TDM pad)
    constexpr int BSTR = BN + 8;   // 136 (256B rows + 16B TDM pad)
    __shared__ u16 As[2][BM][ASTR];
    __shared__ u16 Bs[2][BK][BSTR];

    const int tid  = threadIdx.x;
    const int lane = tid & 31, wave = tid >> 5;
    const int wm = wave & 3;    // 0..3  -> 32 rows each
    const int wn = wave >> 2;   // 0..1  -> 64 cols each
    const int rbase = blockIdx.y * BM;
    const int cbase = blockIdx.x * BN;
    const int hl = lane >> 4, nn = lane & 15;

    floatx8 acc[2][4];
    #pragma unroll
    for (int i = 0; i < 2; ++i)
        #pragma unroll
        for (int j = 0; j < 4; ++j)
            #pragma unroll
            for (int e = 0; e < 8; ++e) acc[i][j][e] = 0.f;

    const int nk = K / BK;

#if USE_TDM
    const u32 ldsA[2] = {(u32)(size_t)&As[0][0][0], (u32)(size_t)&As[1][0][0]};
    const u32 ldsB[2] = {(u32)(size_t)&Bs[0][0][0], (u32)(size_t)&Bs[1][0][0]};
    auto issueTile = [&](int buf, int kt) {
        // A tile: 128 rows x 32 cols, row stride K; 64B rows -> pad codes 3/3
        tdm_load_2d(ldsA[buf], A + (size_t)rbase * K + kt * BK,
                    (u32)K, (u32)M, (u64)K, BK, BM, 3, 3);
        // B tile: 32 rows x 128 cols, row stride N; 256B rows -> pad codes 5/3
        tdm_load_2d(ldsB[buf], Bm + (size_t)(kt * BK) * N + cbase,
                    (u32)N, (u32)K, (u64)N, BN, BK, 5, 3);
    };
    if (wave == 0) issueTile(0, 0);
#else
    uint4 ar[2], br[2];
    auto loadG = [&](int kt) {
        #pragma unroll
        for (int i = 0; i < 2; ++i) {
            int c = tid + i * 256;
            int arow = c >> 2, aseg = c & 3;
            ar[i] = *(const uint4*)(A + (size_t)(rbase + arow) * K + kt * BK + aseg * 8);
            int brow = c >> 4, bseg = c & 15;
            br[i] = *(const uint4*)(Bm + (size_t)(kt * BK + brow) * N + cbase + bseg * 8);
        }
    };
    auto storeL = [&](int buf) {
        #pragma unroll
        for (int i = 0; i < 2; ++i) {
            int c = tid + i * 256;
            int arow = c >> 2, aseg = c & 3;
            *(uint4*)&As[buf][arow][aseg * 8] = ar[i];
            int brow = c >> 4, bseg = c & 15;
            *(uint4*)&Bs[buf][brow][bseg * 8] = br[i];
        }
    };
    loadG(0);
    storeL(0);
    __syncthreads();
#endif

    for (int kt = 0; kt < nk; ++kt) {
        const int buf = kt & 1;
#if USE_TDM
        if (wave == 0) {
            if (kt + 1 < nk) {
                issueTile(buf ^ 1, kt + 1);             // overlap DMA with WMMA
                __builtin_amdgcn_s_wait_tensorcnt(2);   // current tile complete
            } else {
                __builtin_amdgcn_s_wait_tensorcnt(0);
            }
        }
        __syncthreads();   // tile `buf` visible to all waves
#else
        if (kt + 1 < nk) loadG(kt + 1);
#endif

        FragBF af[2], bf_[4];
        #pragma unroll
        for (int tm = 0; tm < 2; ++tm) {
            const int m = wm * 32 + tm * 16 + nn;
            #pragma unroll
            for (int v = 0; v < 8; ++v) {
                const int k = 2 * (v & 3) + (v >> 2) * 16 + hl * 8;
                af[tm].u[v] = *(const u32*)&As[buf][m][k];
            }
        }
        #pragma unroll
        for (int tn = 0; tn < 4; ++tn) {
            const int nb = wn * 64 + tn * 16;
            #pragma unroll
            for (int v = 0; v < 8; ++v)
                bf_[tn].u[v] = (u32)Bs[buf][2 * (v & 3) + (v >> 2) * 16 + hl * 8][nb + nn] | ((u32)Bs[buf][2 * (v & 3) + (v >> 2) * 16 + hl * 8 + 1][nb + nn] << 16);
        }
        #pragma unroll
        for (int tm = 0; tm < 2; ++tm)
            #pragma unroll
            for (int tn = 0; tn < 4; ++tn)
                acc[tm][tn] = __builtin_amdgcn_wmma_f32_16x16x32_bf16(
                    false, af[tm].v, false, bf_[tn].v, (short)0, acc[tm][tn],
                    false, false);

#if USE_TDM
        __syncthreads();   // all reads of `buf` done before its DMA overwrite
#else
        if (kt + 1 < nk) {
            __syncthreads();
            storeL(buf ^ 1);
            __syncthreads();
        }
#endif
    }

    // epilogue (C layout: VGPR r -> row r + 8*(lane>=16), col = lane&15)
    #pragma unroll
    for (int tm = 0; tm < 2; ++tm)
        #pragma unroll
        for (int tn = 0; tn < 4; ++tn)
            #pragma unroll
            for (int r = 0; r < 8; ++r) {
                const int row = rbase + wm * 32 + tm * 16 + hl * 8 + r;
                const int col = cbase + wn * 64 + tn * 16 + nn;
                const float a = acc[tm][tn][r];
                if constexpr (EPI == 0) {
                    outH[(size_t)row * N + col] = f2bf(a);
                } else if constexpr (EPI == 1) {
                    const float x = a + bias[col];
                    const float gl = 0.5f * x * (1.0f + erff(x * 0.70710678118654752f));
                    outH[(size_t)row * N + col] = f2bf(gl);
                } else {
                    outF[(size_t)row * N + col] =
                        a + bias[col] + resid[(size_t)row * N + col];
                }
            }
}

// ---------------------------------------------------------------------------
// Flash attention (no mask): qkv bf16 [B*T, 3*1024] -> y bf16 [B*T, 1024]
// grid = (T/128, NH, B); 8 waves; online softmax over 64-key blocks.
// V block staged by TDM (padded rows); K manually transposed into LDS.
// ---------------------------------------------------------------------------
__global__ __launch_bounds__(256) void attn_kernel(const u16* __restrict__ qkv,
                                                   u16* __restrict__ y) {
    constexpr int D3 = 3072, T = 2048, D = 1024;
    __shared__ u16 Kt[64][72];        // [dh][key]  (transposed K block)
    __shared__ u16 Vs[64][72];        // [key][dh]  (128B rows + 16B TDM pad)
    __shared__ u16 Pl[8][16][72];     // per-wave P tile (16 q-rows x 64 keys)

    const int tid = threadIdx.x, lane = tid & 31, wave = tid >> 5;
    const int qt = blockIdx.x, h = blockIdx.y, b = blockIdx.z;
    const int hl = lane >> 4, nn = lane & 15;

    // Preload Q fragments (A-operand layout), two K=32 steps covering DH=64.
    FragBF qf[2];
    {
        const int qm = qt * 128 + wave * 16 + nn;
        const size_t base = ((size_t)b * T + qm) * D3 + h * 64;
        #pragma unroll
        for (int ks = 0; ks < 2; ++ks)
            #pragma unroll
            for (int v = 0; v < 8; ++v) {
                const int k = 2 * (v & 3) + (v >> 2) * 16 + hl * 8 + ks * 32;
                qf[ks].u[v] = *(const u32*)(qkv + base + k);
            }
    }

#if USE_TDM
    const u32 ldsV = (u32)(size_t)&Vs[0][0];
#endif

    float rmax[8], rsum[8];
    floatx8 yacc[4];
    #pragma unroll
    for (int r = 0; r < 8; ++r) { rmax[r] = -1e30f; rsum[r] = 0.f; }
    #pragma unroll
    for (int dt = 0; dt < 4; ++dt)
        #pragma unroll
        for (int e = 0; e < 8; ++e) yacc[dt][e] = 0.f;

    for (int kb = 0; kb < T / 64; ++kb) {
#if USE_TDM
        // V block 64x64, row stride 3072; 128B rows -> pad codes 4/3
        if (wave == 0)
            tdm_load_2d(ldsV,
                        qkv + ((size_t)b * T + kb * 64) * D3 + h * 64 + 2048,
                        64, 64, (u64)D3, 64, 64, 4, 3);
#endif
        // K block transposed into LDS (manual scatter; TDM cannot transpose)
        #pragma unroll
        for (int i = 0; i < 2; ++i) {
            const int c = tid + i * 256;
            const int key = c >> 3, seg = c & 7;
            const size_t base =
                ((size_t)b * T + kb * 64 + key) * D3 + h * 64 + seg * 8;
            const uint4 kv = *(const uint4*)(qkv + base + 1024);
            const u16* kp = (const u16*)&kv;
            #pragma unroll
            for (int j = 0; j < 8; ++j) Kt[seg * 8 + j][key] = kp[j];
#if !USE_TDM
            *(uint4*)&Vs[key][seg * 8] = *(const uint4*)(qkv + base + 2048);
#endif
        }
#if USE_TDM
        if (wave == 0) __builtin_amdgcn_s_wait_tensorcnt(0);
#endif
        __syncthreads();

        // S = (Q @ K^T) * 1/sqrt(64), four 16x16 key tiles
        floatx8 s[4];
        #pragma unroll
        for (int nt = 0; nt < 4; ++nt) {
            #pragma unroll
            for (int e = 0; e < 8; ++e) s[nt][e] = 0.f;
            #pragma unroll
            for (int ks = 0; ks < 2; ++ks) {
                FragBF bk;
                #pragma unroll
                for (int v = 0; v < 8; ++v)
                    bk.u[v] = (u32)Kt[ks * 32 + 2 * (v & 3) + (v >> 2) * 16 + hl * 8][nt * 16 + nn] | ((u32)Kt[ks * 32 + 2 * (v & 3) + (v >> 2) * 16 + hl * 8 + 1][nt * 16 + nn] << 16);
                s[nt] = __builtin_amdgcn_wmma_f32_16x16x32_bf16(
                    false, qf[ks].v, false, bk.v, (short)0, s[nt], false, false);
            }
        }

        // online softmax (rows live across the 16-lane half-groups)
        float fac[8];
        #pragma unroll
        for (int r = 0; r < 8; ++r) {
            float mx = -1e30f;
            #pragma unroll
            for (int nt = 0; nt < 4; ++nt) {
                s[nt][r] *= 0.125f;
                mx = fmaxf(mx, s[nt][r]);
            }
            #pragma unroll
            for (int m = 8; m >= 1; m >>= 1) mx = fmaxf(mx, __shfl_xor(mx, m, 32));
            const float nm = fmaxf(rmax[r], mx);
            fac[r] = __expf(rmax[r] - nm);
            rmax[r] = nm;
            float ps = 0.f;
            #pragma unroll
            for (int nt = 0; nt < 4; ++nt) {
                const float p = __expf(s[nt][r] - nm);
                s[nt][r] = p;
                ps += p;
            }
            #pragma unroll
            for (int m = 8; m >= 1; m >>= 1) ps += __shfl_xor(ps, m, 32);
            rsum[r] = rsum[r] * fac[r] + ps;
        }
        #pragma unroll
        for (int dt = 0; dt < 4; ++dt)
            #pragma unroll
            for (int r = 0; r < 8; ++r) yacc[dt][r] *= fac[r];

        // stage P (bf16) in per-wave LDS, reload in A-operand layout
        #pragma unroll
        for (int nt = 0; nt < 4; ++nt)
            #pragma unroll
            for (int r = 0; r < 8; ++r)
                Pl[wave][hl * 8 + r][nt * 16 + nn] = f2bf(s[nt][r]);

        #pragma unroll
        for (int ks = 0; ks < 2; ++ks) {
            FragBF ap;
            #pragma unroll
            for (int v = 0; v < 8; ++v) {
                const int k = 2 * (v & 3) + (v >> 2) * 16 + hl * 8 + ks * 32;
                ap.u[v] = *(const u32*)&Pl[wave][nn][k];
            }
            #pragma unroll
            for (int dt = 0; dt < 4; ++dt) {
                FragBF bv;
                #pragma unroll
                for (int v = 0; v < 8; ++v)
                    bv.u[v] = (u32)Vs[ks * 32 + 2 * (v & 3) + (v >> 2) * 16 + hl * 8][dt * 16 + nn] | ((u32)Vs[ks * 32 + 2 * (v & 3) + (v >> 2) * 16 + hl * 8 + 1][dt * 16 + nn] << 16);
                yacc[dt] = __builtin_amdgcn_wmma_f32_16x16x32_bf16(
                    false, ap.v, false, bv.v, (short)0, yacc[dt], false, false);
            }
        }
        __syncthreads();
    }

    // normalize + store
    #pragma unroll
    for (int dt = 0; dt < 4; ++dt)
        #pragma unroll
        for (int r = 0; r < 8; ++r) {
            const int row = qt * 128 + wave * 16 + hl * 8 + r;
            const float ov = yacc[dt][r] / rsum[r];
            y[((size_t)b * T + row) * D + h * 64 + dt * 16 + nn] = f2bf(ov);
        }
}

// ---------------------------------------------------------------------------
// Host-side orchestration
// ---------------------------------------------------------------------------
extern "C" void kernel_launch(void* const* d_in, const int* in_sizes, int n_in,
                              void* d_out, int out_size, void* d_ws, size_t ws_size,
                              hipStream_t stream) {
    (void)in_sizes; (void)n_in; (void)out_size; (void)ws_size;
    constexpr int B = 2, T = 2048, D = 1024, MLP = 4096;
    constexpr int BT = B * T;

    const float* inputs = (const float*)d_in[0];
    const float* ln1_g  = (const float*)d_in[3];
    const float* ln1_b  = (const float*)d_in[4];
    const float* ln2_g  = (const float*)d_in[5];
    const float* ln2_b  = (const float*)d_in[6];
    const float* w_qkv  = (const float*)d_in[7];
    const float* w_out  = (const float*)d_in[8];
    const float* b_out  = (const float*)d_in[9];
    const float* w1     = (const float*)d_in[10];
    const float* b1     = (const float*)d_in[11];
    const float* w2     = (const float*)d_in[12];
    const float* b2     = (const float*)d_in[13];

    char* ws = (char*)d_ws;
    size_t off = 0;
    auto alloc = [&](size_t bytes) { char* p = ws + off; off += (bytes + 255) & ~(size_t)255; return p; };
    u16*   wqkv_h = (u16*)alloc((size_t)D * 3 * D * 2);
    u16*   wout_h = (u16*)alloc((size_t)D * D * 2);
    u16*   w1_h   = (u16*)alloc((size_t)D * MLP * 2);
    u16*   w2_h   = (u16*)alloc((size_t)MLP * D * 2);
    u16*   xn_h   = (u16*)alloc((size_t)BT * D * 2);
    u16*   qkv_h  = (u16*)alloc((size_t)BT * 3 * D * 2);
    u16*   y_h    = (u16*)alloc((size_t)BT * D * 2);
    float* xres_f = (float*)alloc((size_t)BT * D * 4);
    u16*   h_h    = (u16*)alloc((size_t)BT * D * 2);
    u16*   h1_h   = (u16*)alloc((size_t)BT * MLP * 2);

    auto cvt = [&](const float* src, u16* dst, int n) {
        cvt_f32_bf16<<<(n + 255) / 256, 256, 0, stream>>>(src, dst, n);
    };
    cvt(w_qkv, wqkv_h, D * 3 * D);
    cvt(w_out, wout_h, D * D);
    cvt(w1,    w1_h,   D * MLP);
    cvt(w2,    w2_h,   MLP * D);

    // x1 = LN1(inputs) -> bf16
    ln_kernel<<<BT, 256, 0, stream>>>(inputs, ln1_g, ln1_b, xn_h, D);

    // qkv = x1 @ w_qkv  (bf16 out)
    gemm_bf16_kernel<0><<<dim3(3 * D / 128, BT / 128), 256, 0, stream>>>(
        xn_h, wqkv_h, nullptr, nullptr, qkv_h, nullptr, BT, 3 * D, D);

    // flash attention -> y (bf16)
    attn_kernel<<<dim3(T / 128, 16, B), 256, 0, stream>>>(qkv_h, y_h);

    // xres = y @ w_out + b_out + inputs  (fp32)
    gemm_bf16_kernel<2><<<dim3(D / 128, BT / 128), 256, 0, stream>>>(
        y_h, wout_h, b_out, inputs, nullptr, xres_f, BT, D, D);

    // h = LN2(xres) -> bf16
    ln_kernel<<<BT, 256, 0, stream>>>(xres_f, ln2_g, ln2_b, h_h, D);

    // h1 = GELU(h @ w1 + b1)  (bf16)
    gemm_bf16_kernel<1><<<dim3(MLP / 128, BT / 128), 256, 0, stream>>>(
        h_h, w1_h, b1, nullptr, h1_h, nullptr, BT, MLP, D);

    // out = h1 @ w2 + b2 + xres  (fp32)
    gemm_bf16_kernel<2><<<dim3(D / 128, BT / 128), 256, 0, stream>>>(
        h1_h, w2_h, b2, xres_f, nullptr, (float*)d_out, BT, D, MLP);
}
